// BidirectionalConvLSTMSkip_44538810859575
// MI455X (gfx1250) — matrix operand order repaired and checked by reading, not compiled, not run on hardware
//
#include <hip/hip_runtime.h>
#include <stdint.h>

// ---------------------------------------------------------------------------
// BidirectionalConvLSTMSkip for MI455X (gfx1250, wave32, WMMA).
//
// Strategy (see analysis): row-recurrent ConvLSTM as per-row bf16 WMMA GEMMs
// (3 column-shifted K=192 passes emulate the 3-tap width conv), cell state in
// VGPRs, h through LDS, x rows staged via TDM (tensor_load_to_lds) with
// double buffering. Projection = second WMMA GEMM; BN = reduce + fuse passes.
// ---------------------------------------------------------------------------

#define BATCH 8
#define CCH   96
#define HH    192
#define WW    192
#define HCH   96
#define KIN   192          // C + HC
#define NPIX  (HH * WW)    // 36864
#define NTOT  (BATCH * CCH * HH * WW)

typedef __bf16 bf16;
typedef __attribute__((ext_vector_type(16))) bf16  v16bf;
typedef __attribute__((ext_vector_type(8)))  bf16  v8bf;
typedef __attribute__((ext_vector_type(4)))  bf16  v4bf;
typedef __attribute__((ext_vector_type(8)))  float v8f;

// ---- workspace layout (bytes) ---------------------------------------------
#define WS_PACKA   0                      // 2*24*3*6*32*16 bf16 = 884736 B
#define WS_PACKP   884736                 // 6*6*32*16 bf16      = 36864 B
#define WS_XP      921600                 // B*H*W*C bf16        = 56623104 B
#define WS_FEATS   57544704               // B*H*W*192 bf16      = 113246208 B
#define WS_STATS   170790912              // 192 f32 (sum, sumsq)
#define WS_NEEDED  170791680

// ---- LDS layout for the ConvLSTM kernel (bytes) ---------------------------
// x/h image regions are [194 cols][96 chans] bf16, col stride 192 B.
// col 0 and col 193 are permanent zero pads for the width-3 conv.
#define LDS_X0    0
#define LDS_X1    37248
#define LDS_H     74496
#define LDS_BIAS  111744                  // 384 f32
#define LDS_BYTES 113280
#define COLB      192                     // bytes per column (96 bf16)
#define ROWB      36864                   // one x row payload (192 cols * 192 B)

__device__ __forceinline__ v8f wmma_bf16(v16bf a, v16bf b, v8f c) {
  return __builtin_amdgcn_wmma_f32_16x16x32_bf16(false, a, false, b, (short)0,
                                                 c, false, false);
}

__device__ __forceinline__ float sigf(float x) {
  return 1.0f / (1.0f + __expf(-x));
}

// ---------------------------------------------------------------------------
// TDM: 1-D tensor_load_to_lds (global -> LDS), 8-byte elements.
// Descriptor packing per CDNA5 ISA D# groups 0/1.
// ---------------------------------------------------------------------------
#if __has_builtin(__builtin_amdgcn_tensor_load_to_lds) && \
    __has_builtin(__builtin_amdgcn_s_wait_tensorcnt)
#define HAVE_TDM 1
typedef unsigned int v4u __attribute__((ext_vector_type(4)));
typedef int          v4i __attribute__((ext_vector_type(4)));
typedef int          v8i __attribute__((ext_vector_type(8)));

__device__ __forceinline__ unsigned lds_off_of(const void* p) {
  // LDS aperture addresses carry the byte offset in ADDR[31:0].
  return (unsigned)(unsigned long long)(uintptr_t)p;
}

__device__ __forceinline__ void tdm_load_1d(const void* gsrc, unsigned lds_addr,
                                            unsigned nbytes) {
  unsigned long long ga = (unsigned long long)(uintptr_t)gsrc;
  unsigned n8 = nbytes >> 3;                       // 8-byte elements
  v4u g0;
  g0[0] = 1u;                                      // count=1 user descriptor
  g0[1] = lds_addr;                                // lds_addr (bytes)
  g0[2] = (unsigned)(ga & 0xffffffffu);            // global_addr[31:0]
  g0[3] = (unsigned)((ga >> 32) & 0x01ffffffu) | (2u << 30);  // [56:32]|type=2
  v8i g1;
  g1[0] = (int)(3u << 16);                         // data_size=8B, mask=0
  g1[1] = (int)((n8 & 0xffffu) << 16);             // tensor_dim0[15:0]
  g1[2] = (int)(((n8 >> 16) & 0xffffu) | (1u << 16)); // dim0 hi | tensor_dim1=1
  g1[3] = (int)((n8 & 0xffffu) << 16);             // tile_dim0
  g1[4] = 0;                                       // tile_dim1/2 unused (1-D)
  g1[5] = (int)n8;                                 // tensor_dim0_stride lo
  g1[6] = 0;
  g1[7] = 0;
  v4i z4 = {0, 0, 0, 0};
#if __clang_major__ >= 23
  v8i z8 = {0, 0, 0, 0, 0, 0, 0, 0};
  __builtin_amdgcn_tensor_load_to_lds(g0, g1, z4, z4, z8, 0);
#else
  __builtin_amdgcn_tensor_load_to_lds(g0, g1, z4, z4, 0);
#endif
}
#else
#define HAVE_TDM 0
#endif

// ---------------------------------------------------------------------------
// Prep kernel A: pack gate weights (middle kernel row) and w_proj into the
// A-fragment layout (16x32 bf16) in this code's contiguous k order, matching the B loads: lane l holds row M = l%16 and
// K = kk*32 + (l/16)*16 + e.
// ---------------------------------------------------------------------------
#define NPACKA (2 * 24 * 3 * 6 * 32 * 16)   // 442368
#define NPACKP (6 * 6 * 32 * 16)            // 18432

__global__ void k_pack_w(const float* __restrict__ wf,
                         const float* __restrict__ wb,
                         const float* __restrict__ wproj,
                         bf16* __restrict__ packA, bf16* __restrict__ packP) {
  int i = blockIdx.x * 256 + threadIdx.x;
  if (i < NPACKA) {
    int e    = i & 15;
    int l    = (i >> 4) & 31;
    int kk   = (i >> 9) % 6;
    int tap  = (i / (16 * 32 * 6)) % 3;
    int mt   = (i / (16 * 32 * 6 * 3)) % 24;
    int dir  = i / (16 * 32 * 6 * 3 * 24);
    int m = mt * 16 + (l & 15);
    int k = kk * 32 + ((l >> 4) << 4) + e;
    const float* w = dir ? wb : wf;
    // w shape (384, 192, 3, 3); middle height row (index 1), width tap.
    packA[i] = (bf16)w[((m * KIN + k) * 3 + 1) * 3 + tap];
  } else if (i < NPACKA + NPACKP) {
    int j  = i - NPACKA;
    int e  = j & 15;
    int l  = (j >> 4) & 31;
    int kk = (j >> 9) % 6;
    int mt = j / (16 * 32 * 6);
    int m = mt * 16 + (l & 15);
    int k = kk * 32 + ((l >> 4) << 4) + e;
    packP[j] = (bf16)wproj[m * KIN + k];
  }
}

// ---------------------------------------------------------------------------
// Prep kernel B: x (B,C,H,W) f32 -> xp (B,H,W,C) bf16 so each row is one
// contiguous 36864-B block the TDM can copy in a single 1-D descriptor.
// ---------------------------------------------------------------------------
__global__ void k_pack_x(const float* __restrict__ x, bf16* __restrict__ xp) {
  int i = blockIdx.x * 256 + threadIdx.x;    // over B*H*W * 24 groups of 4 ch
  if (i >= BATCH * HH * WW * 24) return;
  int cg = i % 24;
  int n  = i / 24;                           // (b*H + h)*W + w
  int w  = n % WW;
  int bh = n / WW;
  int h  = bh % HH;
  int b  = bh / HH;
  v4bf v;
#pragma unroll
  for (int j = 0; j < 4; ++j) {
    int c = cg * 4 + j;
    v[j] = (bf16)x[((size_t)(b * CCH + c) * HH + h) * WW + w];
  }
  *(v4bf*)(xp + (size_t)n * CCH + cg * 4) = v;
}

// ---------------------------------------------------------------------------
// Kernel 1: bidirectional ConvLSTM row scan.
// grid = 16 blocks (dir*8 + batch), block = 768 threads (24 waves).
// Wave w owns 3 (hct, nt) pairs; per pair it computes all 4 gate 16x16 tiles
// so i/f/g/o for the same (channel, w) land in the same lane -> the cell
// state c lives in 24 VGPRs per wave for the whole scan.
// ---------------------------------------------------------------------------
__global__ void __launch_bounds__(768, 1)
k1_convlstm(const bf16* __restrict__ xp, const bf16* __restrict__ packA,
            const float* __restrict__ bias_f, const float* __restrict__ bias_b,
            bf16* __restrict__ feats) {
  extern __shared__ __align__(32) char smem[];
  const int tid   = threadIdx.x;
  const int wave  = tid >> 5;
  const int lane  = tid & 31;
  const int nlo   = lane & 15;           // column within the 16x16 tile
  const int halfM = (lane >> 4) << 3;    // C/D row half (M += 8 for lanes>=16)
  const int khalf = (lane >> 4) << 4;    // B-fragment K half (bytes/2)
  const int dir   = blockIdx.x >> 3;
  const int bb    = blockIdx.x & 7;

  // ---- zero x pads + h region, load bias into LDS -------------------------
  for (int i = tid; i < LDS_BIAS / 16; i += 768) {
    int4 z = {0, 0, 0, 0};
    *(int4*)(smem + i * 16) = z;
  }
  {
    const float* bp = dir ? bias_b : bias_f;
    if (tid < 384) ((float*)(smem + LDS_BIAS))[tid] = bp[tid];
  }
  __syncthreads();

  const bf16* xrow0 =
      xp + (size_t)(bb * HH + (dir ? HH - 1 : 0)) * WW * CCH;
#if HAVE_TDM
  if (wave == 0) {
    tdm_load_1d(xrow0, lds_off_of(smem + LDS_X0 + COLB), ROWB);
    __builtin_amdgcn_s_wait_tensorcnt(0);
  }
#else
  for (int i = tid; i < ROWB / 16; i += 768)
    *(int4*)(smem + LDS_X0 + COLB + i * 16) =
        *(const int4*)((const char*)xrow0 + i * 16);
#endif
  __syncthreads();

  float cst[3][8];
#pragma unroll
  for (int p = 0; p < 3; ++p)
#pragma unroll
    for (int r = 0; r < 8; ++r) cst[p][r] = 0.0f;

  for (int step = 0; step < HH; ++step) {
    const int row = dir ? (HH - 1 - step) : step;
    const int cur = step & 1;
    // ---- kick off DMA of the next row into the other x buffer -------------
    if (step + 1 < HH) {
      const int nrow = dir ? (HH - 2 - step) : (step + 1);
      const bf16* src = xp + (size_t)(bb * HH + nrow) * WW * CCH;
      char* dst = smem + (cur ? LDS_X0 : LDS_X1) + COLB;
#if HAVE_TDM
      if (wave == 0) tdm_load_1d(src, lds_off_of(dst), ROWB);
#else
      for (int i = tid; i < ROWB / 16; i += 768)
        *(int4*)(dst + i * 16) = *(const int4*)((const char*)src + i * 16);
#endif
    }

    const char* xbase = smem + (cur ? LDS_X1 : LDS_X0);
    const char* hbase = smem + LDS_H;
    v8bf hsave[3];

#pragma unroll
    for (int p = 0; p < 3; ++p) {
      const int pairIdx = wave * 3 + p;
      const int hct = pairIdx / 12;
      const int nt  = pairIdx % 12;
      const int wcol = nt * 16 + nlo;                 // 0..191

      v8f acc[4];
#pragma unroll
      for (int g = 0; g < 4; ++g)
        acc[g] = *(const v8f*)(smem + LDS_BIAS +
                               (g * HCH + hct * 16 + halfM) * 4);

#pragma unroll 1
      for (int tap = 0; tap < 3; ++tap) {
        const int pcol = wcol + tap;                  // padded col 0..193
        // -- K 0..95 from the x image --
#pragma unroll 1
        for (int kk = 0; kk < 3; ++kk) {
          v16bf bfr =
              *(const v16bf*)(xbase + pcol * COLB + (kk * 32 + khalf) * 2);
#pragma unroll
          for (int g = 0; g < 4; ++g) {
            const int mt = g * 6 + hct;
            v16bf afr = *(const v16bf*)(
                packA +
                ((size_t)(((dir * 24 + mt) * 3 + tap) * 6 + kk) * 32 + lane) *
                    16);
            acc[g] = wmma_bf16(afr, bfr, acc[g]);
          }
        }
        // -- K 96..191 from the h image --
#pragma unroll 1
        for (int kk = 0; kk < 3; ++kk) {
          v16bf bfr =
              *(const v16bf*)(hbase + pcol * COLB + (kk * 32 + khalf) * 2);
#pragma unroll
          for (int g = 0; g < 4; ++g) {
            const int mt = g * 6 + hct;
            v16bf afr = *(const v16bf*)(
                packA +
                ((size_t)(((dir * 24 + mt) * 3 + tap) * 6 + (kk + 3)) * 32 +
                 lane) *
                    16);
            acc[g] = wmma_bf16(afr, bfr, acc[g]);
          }
        }
      }

      // ---- LSTM pointwise update (c in registers) -------------------------
      v8bf hv;
#pragma unroll
      for (int r = 0; r < 8; ++r) {
        float ig = sigf(acc[0][r]);
        float fg = sigf(acc[1][r]);
        float gg = tanhf(acc[2][r]);
        float og = sigf(acc[3][r]);
        float cn = fg * cst[p][r] + ig * gg;
        cst[p][r] = cn;
        hv[r] = (bf16)(og * tanhf(cn));
      }
      hsave[p] = hv;
      // ---- emit features (b,h,w, dir*96+hc) bf16 --------------------------
      size_t n = ((size_t)(bb * HH + row)) * WW + wcol;
      *(v8bf*)(feats + n * KIN + dir * HCH + hct * 16 + halfM) = hv;
    }

    __syncthreads();   // everyone done reading old h
#pragma unroll
    for (int p = 0; p < 3; ++p) {
      const int pairIdx = wave * 3 + p;
      const int hct = pairIdx / 12;
      const int nt  = pairIdx % 12;
      const int pcol = 1 + nt * 16 + nlo;
      *(v8bf*)(smem + LDS_H + pcol * COLB + (hct * 16 + halfM) * 2) = hsave[p];
    }
#if HAVE_TDM
    if (wave == 0 && step + 1 < HH) __builtin_amdgcn_s_wait_tensorcnt(0);
#endif
    __syncthreads();   // new h + next x row visible
  }
}

// ---------------------------------------------------------------------------
// Kernel 2: y = w_proj @ feats  (M=96, K=192, N=B*H*W=294912), WMMA bf16.
// One wave per 16-column tile; B fragment reused across the 6 M tiles.
// ---------------------------------------------------------------------------
__global__ void __launch_bounds__(256)
k2_proj(const bf16* __restrict__ feats, const bf16* __restrict__ packP,
        float* __restrict__ y) {
  const int wave  = threadIdx.x >> 5;
  const int lane  = threadIdx.x & 31;
  const int nlo   = lane & 15;
  const int halfM = (lane >> 4) << 3;
  const int khalf = (lane >> 4) << 4;
  const int nt = blockIdx.x * 8 + wave;
  const int n  = nt * 16 + nlo;

  v8f acc[6];
#pragma unroll
  for (int mt = 0; mt < 6; ++mt) { v8f z = {}; acc[mt] = z; }

  const bf16* colp = feats + (size_t)n * KIN;
#pragma unroll 1
  for (int kk = 0; kk < 6; ++kk) {
    v16bf bfr = *(const v16bf*)(colp + kk * 32 + khalf);
#pragma unroll
    for (int mt = 0; mt < 6; ++mt) {
      v16bf afr = *(const v16bf*)(packP + ((mt * 6 + kk) * 32 + lane) * 16);
      acc[mt] = wmma_bf16(afr, bfr, acc[mt]);
    }
  }

  const int b  = n / NPIX;
  const int hw = n % NPIX;
  float* outb = y + (size_t)b * CCH * NPIX + hw;
#pragma unroll
  for (int mt = 0; mt < 6; ++mt)
#pragma unroll
    for (int r = 0; r < 8; ++r)
      outb[(size_t)(mt * 16 + halfM + r) * NPIX] = acc[mt][r];
}

// ---------------------------------------------------------------------------
// Kernel 3: per-channel sum / sumsq of y (one block per (b,o) plane).
// ---------------------------------------------------------------------------
__global__ void k3_stats(const float* __restrict__ y, float* __restrict__ st) {
  __shared__ float ssum[256], ssq[256];
  const int o = blockIdx.x % CCH;
  const float* plane = y + (size_t)blockIdx.x * NPIX;
  float s = 0.f, q = 0.f;
  for (int i = threadIdx.x; i < NPIX; i += 256) {
    float v = plane[i];
    s += v;
    q += v * v;
  }
  ssum[threadIdx.x] = s;
  ssq[threadIdx.x] = q;
  __syncthreads();
  for (int off = 128; off; off >>= 1) {
    if (threadIdx.x < off) {
      ssum[threadIdx.x] += ssum[threadIdx.x + off];
      ssq[threadIdx.x]  += ssq[threadIdx.x + off];
    }
    __syncthreads();
  }
  if (threadIdx.x == 0) {
    atomicAdd(&st[o], ssum[0]);
    atomicAdd(&st[CCH + o], ssq[0]);
  }
}

// ---------------------------------------------------------------------------
// Kernel 4: BatchNorm(train) + ReLU + skip, in place on y (= d_out).
// ---------------------------------------------------------------------------
__global__ void k4_bn(float* __restrict__ y, const float* __restrict__ x,
                      const float* __restrict__ st,
                      const float* __restrict__ gamma,
                      const float* __restrict__ beta) {
  size_t i = (size_t)blockIdx.x * 256 + threadIdx.x;
  if (i >= (size_t)NTOT) return;
  int o = (int)((i / NPIX) % CCH);
  const float N = (float)(BATCH * NPIX);
  float mu  = st[o] / N;
  float var = st[CCH + o] / N - mu * mu;
  float v = gamma[o] * (y[i] - mu) * rsqrtf(var + 1e-5f) + beta[o];
  v = fmaxf(v, 0.0f);
  y[i] = v + x[i];
}

// ---------------------------------------------------------------------------
extern "C" void kernel_launch(void* const* d_in, const int* in_sizes, int n_in,
                              void* d_out, int out_size, void* d_ws,
                              size_t ws_size, hipStream_t stream) {
  const float* x      = (const float*)d_in[0];
  const float* w_f    = (const float*)d_in[1];
  const float* b_f    = (const float*)d_in[2];
  const float* w_b    = (const float*)d_in[3];
  const float* b_b    = (const float*)d_in[4];
  const float* w_proj = (const float*)d_in[5];
  const float* gamma  = (const float*)d_in[6];
  const float* beta   = (const float*)d_in[7];

  char* ws = (char*)d_ws;
  bf16*  packA = (bf16*)(ws + WS_PACKA);
  bf16*  packP = (bf16*)(ws + WS_PACKP);
  bf16*  xp    = (bf16*)(ws + WS_XP);
  bf16*  feats = (bf16*)(ws + WS_FEATS);
  float* stats = (float*)(ws + WS_STATS);
  float* y = (float*)d_out;

  (void)in_sizes; (void)n_in; (void)out_size; (void)ws_size;

  hipFuncSetAttribute((const void*)k1_convlstm,
                      hipFuncAttributeMaxDynamicSharedMemorySize, LDS_BYTES);

  k_pack_w<<<(NPACKA + NPACKP) / 256, 256, 0, stream>>>(w_f, w_b, w_proj,
                                                        packA, packP);
  k_pack_x<<<(BATCH * HH * WW * 24) / 256, 256, 0, stream>>>(x, xp);
  k1_convlstm<<<16, 768, LDS_BYTES, stream>>>(xp, packA, b_f, b_b, feats);
  k2_proj<<<(BATCH * NPIX / 16) / 8, 256, 0, stream>>>(feats, packP, y);
  hipMemsetAsync(stats, 0, 2 * CCH * sizeof(float), stream);
  k3_stats<<<BATCH * CCH, 256, 0, stream>>>(y, stats);
  k4_bn<<<(NTOT + 255) / 256, 256, 0, stream>>>(y, x, stats, gamma, beta);
}
